// LightweightS6Layer_91268055040526
// MI455X (gfx1250) — compile-verified, matrix operand order checked by reading, not run on hardware
//
#include <hip/hip_runtime.h>
#include <hip/hip_bf16.h>

// ---------------------------------------------------------------------------
// LightweightS6Layer for MI455X (gfx1250, wave32, WMMA)
//
//   xz = x @ W_in^T ; xi = silu(xz[:, :768]) ; zs = silu(xz[:, 768:])
//   dt = clip(xi @ W_dt^T + b_dt, 1e-3, 0.1)
//   g[l,d] = sum_s dt[l,s] * exp(clip(dt[l,s] * A[s,d], -10, 10)),  A = -exp(clip(A_log,-10,2))
//   y = cumsum_L(xi * g) + xi * clip(Dp,-10,10) ; y *= zs
//   out = y @ W_out^T
//
// GEMMs run as bf16x3-split WMMA (v_wmma_f32_16x16x32_bf16) for fp32-class
// accuracy; the scan is a chunked (2-pass) prefix sum along L.
// ---------------------------------------------------------------------------

#define B_    4
#define L_    2048
#define DM    512
#define DI    768
#define DS    8
#define MROWS (B_ * L_)   // 8192
#define DT_MIN_ 0.001f
#define DT_MAX_ 0.1f

typedef __attribute__((ext_vector_type(16))) __bf16         v16bf;
typedef __attribute__((ext_vector_type(8)))  float          v8f;
typedef __attribute__((ext_vector_type(8)))  unsigned short ushort8;

union FragBF { v16bf v; unsigned short u[16]; };

__device__ __forceinline__ unsigned short f2bf_rne(float f) {
  unsigned int u = __float_as_uint(f);
  unsigned int r = u + 0x7FFFu + ((u >> 16) & 1u);   // round-to-nearest-even
  return (unsigned short)(r >> 16);
}
__device__ __forceinline__ float bf2f(unsigned short h) {
  return __uint_as_float(((unsigned int)h) << 16);
}
__device__ __forceinline__ float silu_f(float v) { return v / (1.0f + __expf(-v)); }
__device__ __forceinline__ float clampf(float v, float lo, float hi) {
  return fminf(fmaxf(v, lo), hi);
}

// --------------------------- fp32 -> bf16 hi/lo split -----------------------
__global__ __launch_bounds__(256) void split_bf16_kernel(
    const float* __restrict__ src, unsigned short* __restrict__ hi,
    unsigned short* __restrict__ lo, int n) {
  int i = blockIdx.x * blockDim.x + threadIdx.x;
  if (i >= n) return;
  float v = src[i];
  unsigned short h = f2bf_rne(v);
  hi[i] = h;
  lo[i] = f2bf_rne(v - bf2f(h));
}

// ------------------------- WMMA fragment loader -----------------------------
// CDNA5 16-bit A layout (16x32, MxK): lane = M (mod 16); lanes 0-15 hold
// K in {0..7, 16..23}, lanes 16-31 hold K in {8..15, 24..31} -> two
// contiguous 8-element (16B) runs per lane. B (32x16, KxN) from row-major
// W[N,K] is symmetric: lane&15 selects N-row of W, same K pattern.
__device__ __forceinline__ v16bf load_frag(const unsigned short* __restrict__ base,
                                           int ld, int row, int k0, int half) {
  const unsigned short* p = base + (size_t)row * ld + k0 + half * 8;
  ushort8 a = *(const ushort8*)(p);        // K = k0 + half*8 .. +7
  ushort8 b = *(const ushort8*)(p + 16);   // K = k0 + 16 + half*8 .. +7
  FragBF f;
#pragma unroll
  for (int i = 0; i < 8; ++i) { f.u[i] = a[i]; f.u[8 + i] = b[i]; }
  return f.v;
}

// ---------------------- bf16x3-split WMMA GEMM ------------------------------
// C[M,N] = A[M,K] * W[N,K]^T with A,W stored as bf16 (hi, lo) row-major.
// Wave tile: 16M x 64N (4 x v8f accum). Block: 8 waves = 128M x 64N.
// mode 0: plain fp32 store to outA (ld = N)
// mode 1: fused SiLU; cols < DI -> outA (xi, ld=DI), cols >= DI -> outB (zs)
__global__ __launch_bounds__(256) void gemm_bf16x3_kernel(
    const unsigned short* __restrict__ Ahi, const unsigned short* __restrict__ Alo,
    const unsigned short* __restrict__ Bhi, const unsigned short* __restrict__ Blo,
    float* __restrict__ outA, float* __restrict__ outB,
    int M, int N, int K, int mode) {
  const int lane = threadIdx.x & 31;
  const int wave = threadIdx.x >> 5;
  const int half = lane >> 4;
  const int mr   = lane & 15;

  const int m0 = blockIdx.x * 128 + wave * 16;
  const int n0 = blockIdx.y * 64;
  if (m0 >= M) return;

  v8f acc[4] = {};
  const int arow = m0 + mr;

  for (int k0 = 0; k0 < K; k0 += 32) {
    v16bf ah = load_frag(Ahi, K, arow, k0, half);
    v16bf al = load_frag(Alo, K, arow, k0, half);
    if (k0 + 32 < K)  // warm next A k-slab (global_prefetch_b8)
      __builtin_prefetch(Ahi + (size_t)arow * K + k0 + 32, 0, 0);
#pragma unroll
    for (int j = 0; j < 4; ++j) {
      const int brow = n0 + j * 16 + mr;
      v16bf bh = load_frag(Bhi, K, brow, k0, half);
      v16bf bl = load_frag(Blo, K, brow, k0, half);
      acc[j] = __builtin_amdgcn_wmma_f32_16x16x32_bf16(false, ah, false, bh,
                                                       (short)0, acc[j], false, false);
      acc[j] = __builtin_amdgcn_wmma_f32_16x16x32_bf16(false, ah, false, bl,
                                                       (short)0, acc[j], false, false);
      acc[j] = __builtin_amdgcn_wmma_f32_16x16x32_bf16(false, al, false, bh,
                                                       (short)0, acc[j], false, false);
    }
  }

  // D layout: VGPR r -> M = r + 8*half ; lane&15 -> N
#pragma unroll
  for (int j = 0; j < 4; ++j) {
    const int col = n0 + j * 16 + mr;
#pragma unroll
    for (int r = 0; r < 8; ++r) {
      const int m = m0 + r + half * 8;
      float v = acc[j][r];
      if (mode == 0) {
        outA[(size_t)m * N + col] = v;
      } else {
        float s = silu_f(v);
        if (col < DI) outA[(size_t)m * DI + col]        = s;
        else          outB[(size_t)m * DI + (col - DI)] = s;
      }
    }
  }
}

// ------------------------- dt projection (N=8) ------------------------------
// One wave32 per row: 8 dot-products of length 768, cross-lane reduce.
__global__ __launch_bounds__(256) void dt_kernel(
    const float* __restrict__ xi, const float* __restrict__ W_dt,
    const float* __restrict__ b_dt, float* __restrict__ dt) {
  const int gw   = (blockIdx.x * blockDim.x + threadIdx.x) >> 5;
  const int lane = threadIdx.x & 31;
  if (gw >= MROWS) return;
  const float* xr = xi + (size_t)gw * DI;
  float acc[DS] = {};
  for (int k = lane; k < DI; k += 32) {
    float xv = xr[k];
#pragma unroll
    for (int s = 0; s < DS; ++s) acc[s] += xv * W_dt[s * DI + k];
  }
#pragma unroll
  for (int s = 0; s < DS; ++s) {
    float v = acc[s];
#pragma unroll
    for (int off = 16; off > 0; off >>= 1) v += __shfl_xor(v, off, 32);
    if (lane == 0)
      dt[(size_t)gw * DS + s] = clampf(v + b_dt[s], DT_MIN_, DT_MAX_);
  }
}

// -------------------- scan pass 1: u + per-chunk partials -------------------
// grid = (Lchunks=8, Dchunks=3, B=4); thread owns one d-channel of one chunk.
__global__ __launch_bounds__(256) void scan_partial_kernel(
    const float* __restrict__ xi, const float* __restrict__ dt,
    const float* __restrict__ A_log, float* __restrict__ u,
    float* __restrict__ partial) {
  const int lc = blockIdx.x, dc = blockIdx.y, b = blockIdx.z;
  const int d = dc * 256 + threadIdx.x;

  float a[DS];
#pragma unroll
  for (int s = 0; s < DS; ++s)
    a[s] = -__expf(clampf(A_log[s * DI + d], -10.f, 2.f));

  float acc = 0.f;
  const int lbeg = lc * 256;
  for (int l = lbeg; l < lbeg + 256; ++l) {
    const size_t row = (size_t)b * L_ + l;
    const float* dtp = dt + row * DS;
    float g = 0.f;
#pragma unroll
    for (int s = 0; s < DS; ++s) {
      float dts = dtp[s];
      g += dts * __expf(clampf(dts * a[s], -10.f, 10.f));
    }
    float uv = xi[row * DI + d] * g;
    u[row * DI + d] = uv;
    acc += uv;
  }
  partial[((size_t)b * 8 + lc) * DI + d] = acc;
}

// -------- scan pass 2: apply chunk prefix, gate, emit y as split-bf16 -------
__global__ __launch_bounds__(256) void scan_apply_kernel(
    const float* __restrict__ u, const float* __restrict__ xi,
    const float* __restrict__ zs, const float* __restrict__ Dp,
    const float* __restrict__ partial,
    unsigned short* __restrict__ y_hi, unsigned short* __restrict__ y_lo) {
  const int lc = blockIdx.x, dc = blockIdx.y, b = blockIdx.z;
  const int d = dc * 256 + threadIdx.x;
  const float dp = clampf(Dp[d], -10.f, 10.f);

  float acc = 0.f;
  for (int c = 0; c < lc; ++c)
    acc += partial[((size_t)b * 8 + c) * DI + d];

  const int lbeg = lc * 256;
  for (int l = lbeg; l < lbeg + 256; ++l) {
    const size_t idx = ((size_t)b * L_ + l) * DI + d;
    acc += u[idx];
    float y = (acc + xi[idx] * dp) * zs[idx];
    unsigned short h = f2bf_rne(y);
    y_hi[idx] = h;
    y_lo[idx] = f2bf_rne(y - bf2f(h));
  }
}

// ---------------------------------------------------------------------------
extern "C" void kernel_launch(void* const* d_in, const int* in_sizes, int n_in,
                              void* d_out, int out_size, void* d_ws, size_t ws_size,
                              hipStream_t stream) {
  (void)in_sizes; (void)n_in; (void)out_size; (void)ws_size;
  const float* x     = (const float*)d_in[0];
  const float* W_in  = (const float*)d_in[1];
  const float* W_dt  = (const float*)d_in[2];
  const float* b_dt  = (const float*)d_in[3];
  const float* A_log = (const float*)d_in[4];
  const float* Dp    = (const float*)d_in[5];
  const float* W_out = (const float*)d_in[6];
  float* out = (float*)d_out;

  char* wp = (char*)d_ws;
  auto carve = [&](size_t bytes) -> void* {
    void* p = (void*)wp;
    wp += (bytes + 255) & ~(size_t)255;
    return p;
  };

  unsigned short* x_hi   = (unsigned short*)carve((size_t)MROWS * DM * 2);
  unsigned short* x_lo   = (unsigned short*)carve((size_t)MROWS * DM * 2);
  unsigned short* win_hi = (unsigned short*)carve((size_t)2 * DI * DM * 2);
  unsigned short* win_lo = (unsigned short*)carve((size_t)2 * DI * DM * 2);
  unsigned short* wout_hi= (unsigned short*)carve((size_t)DM * DI * 2);
  unsigned short* wout_lo= (unsigned short*)carve((size_t)DM * DI * 2);
  float* xi      = (float*)carve((size_t)MROWS * DI * 4);
  float* zs      = (float*)carve((size_t)MROWS * DI * 4);
  float* dtbuf   = (float*)carve((size_t)MROWS * DS * 4);
  float* ubuf    = (float*)carve((size_t)MROWS * DI * 4);
  float* partial = (float*)carve((size_t)B_ * 8 * DI * 4);
  unsigned short* y_hi = (unsigned short*)carve((size_t)MROWS * DI * 2);
  unsigned short* y_lo = (unsigned short*)carve((size_t)MROWS * DI * 2);

  // 1) fp32 -> bf16 hi/lo splits (x, W_in, W_out)
  {
    int n = MROWS * DM;
    split_bf16_kernel<<<(n + 255) / 256, 256, 0, stream>>>(x, x_hi, x_lo, n);
  }
  {
    int n = 2 * DI * DM;
    split_bf16_kernel<<<(n + 255) / 256, 256, 0, stream>>>(W_in, win_hi, win_lo, n);
  }
  {
    int n = DM * DI;
    split_bf16_kernel<<<(n + 255) / 256, 256, 0, stream>>>(W_out, wout_hi, wout_lo, n);
  }

  // 2) GEMM1 (8192 x 1536 x 512) with fused SiLU + xi/z split
  gemm_bf16x3_kernel<<<dim3(MROWS / 128, (2 * DI) / 64), 256, 0, stream>>>(
      x_hi, x_lo, win_hi, win_lo, xi, zs, MROWS, 2 * DI, DM, 1);

  // 3) dt projection (wave32 per row)
  dt_kernel<<<(MROWS * 32) / 256, 256, 0, stream>>>(xi, W_dt, b_dt, dtbuf);

  // 4) chunked prefix scan along L (state dim folded into per-element gain g)
  scan_partial_kernel<<<dim3(8, 3, 4), 256, 0, stream>>>(xi, dtbuf, A_log, ubuf, partial);
  scan_apply_kernel<<<dim3(8, 3, 4), 256, 0, stream>>>(ubuf, xi, zs, Dp, partial, y_hi, y_lo);

  // 5) GEMM2 (8192 x 512 x 768) -> out
  gemm_bf16x3_kernel<<<dim3(MROWS / 128, DM / 64), 256, 0, stream>>>(
      y_hi, y_lo, wout_hi, wout_lo, out, nullptr, MROWS, DM, DI, 0);
}
